// BasicMambaBlock_66589172957422
// MI455X (gfx1250) — matrix operand order repaired and checked by reading, not compiled, not run on hardware
//
#include <hip/hip_runtime.h>
#include <hip/hip_bf16.h>
#include <math.h>

// ---------------------------------------------------------------------------
// Mamba block on MI455X (gfx1250, wave32).
// GEMMs: bf16 operands, v_wmma_f32_16x16x32_bf16, f32 acc.
// Block tile 128x128, 8 waves (4m x 2n), wave tile 32x64 = 8 WMMA subtiles.
// Staging via GLOBAL_LOAD_ASYNC_TO_LDS_B128 (ASYNCcnt) into double-buffered
// LDS; weights pre-transposed (Wt[n][k]) during the one-time bf16 convert so
// both tiles are straight row-major async copies (no pack/transpose VALU).
// Scan: lane-per-state (16 lanes/channel), shfl_xor reduction.
// ---------------------------------------------------------------------------

typedef __attribute__((ext_vector_type(16))) __bf16 v16bf;
typedef __attribute__((ext_vector_type(8)))  float  v8f;

constexpr int BATCH  = 2;
constexpr int SEQ    = 2048;
constexpr int DM     = 1024;            // D_MODEL
constexpr int DI     = 2048;            // D_INNER
constexpr int DS     = 16;              // D_STATE
constexpr int DTR    = 64;              // DT_RANK
constexpr int DFF    = 4096;            // FF_INNER
constexpr int DCONV  = 4;
constexpr int M_ROWS = BATCH * SEQ;     // 4096 token rows

__device__ __forceinline__ float siluf(float x) { return x / (1.f + __expf(-x)); }
__device__ __forceinline__ float geluf(float x) {
  float x3 = x * x * x;
  return 0.5f * x * (1.f + tanhf(0.7978845608028654f * (x + 0.044715f * x3)));
}
__device__ __forceinline__ float softplusf(float x) {
  return (x > 20.f) ? x : log1pf(__expf(x));
}

// load a 32-byte WMMA fragment as two b128 LDS reads
__device__ __forceinline__ v16bf ld_frag(const __bf16* base, int q0, int q1) {
  union { v16bf v; uint4 q[2]; } u;
  const uint4* p = (const uint4*)base;
  u.q[0] = p[q0];
  u.q[1] = p[q1];
  return u.v;
}

// ---------------------------------------------------------------------------
// C[M,N] = epi( A[M,K](lda,bf16) * Wt[N,K](bf16, row=output col) )
// EPI: 0 none | 1 +bias | 2 softplus(x+bias) | 3 +resid | 4 +bias+resid
// NGUARD: epilogue column bounds check (N=96 GEMM; Wt rows padded for loads).
// Requires: M % 128 == 0, K % 32 == 0, Wt rows padded to mult of 128.
// ---------------------------------------------------------------------------
template <int EPI, bool NGUARD>
__global__ __launch_bounds__(256)
void gemm_bf16_wmma(const __bf16* __restrict__ A, const __bf16* __restrict__ Wt,
                    const float* __restrict__ bias, const float* __restrict__ resid,
                    float* __restrict__ C, int M, int N, int K, int lda, int ldc) {
  constexpr int BM = 128, BN = 128, BK = 32, LDS_K = BK + 8;  // 40 (pad)
  __shared__ __bf16 As[2][BM][LDS_K];   // [m][k], double buffered
  __shared__ __bf16 Bs[2][BN][LDS_K];   // [n][k], double buffered

  const int m0   = blockIdx.y * BM;
  const int n0   = blockIdx.x * BN;
  const int tid  = threadIdx.x;
  const int wave = tid >> 5;
  const int lane = tid & 31;
  const int wm   = wave >> 1;        // 0..3 -> 32-row slab
  const int wn   = wave & 1;         // 0..1 -> 64-col slab
  const int l16  = lane & 15;
  const int lhalf = lane >> 4;       // 0 or 1

  v8f acc[2][4];
  #pragma unroll
  for (int i = 0; i < 2; ++i)
    #pragma unroll
    for (int j = 0; j < 4; ++j)
      #pragma unroll
      for (int v = 0; v < 8; ++v) acc[i][j][v] = 0.f;

  // staging map: 256 threads, each copies 16 bf16 (two b128) of A and of Wt
  const int sr = tid >> 1;            // 0..127 (row of both tiles)
  const int sc = (tid & 1) * 16;      // 0 | 16 (K offset)
  const __bf16* aptr = A  + (size_t)(m0 + sr) * lda + sc;
  const __bf16* wptr = Wt + (size_t)(n0 + sr) * K  + sc;

  // DMA tile (k0) into LDS buffer p: per-lane async gather, ASYNCcnt-tracked.
  auto stage_async = [&](int k0, int p) {
    const unsigned la = (unsigned)(uintptr_t)&As[p][sr][sc];
    const unsigned lb = (unsigned)(uintptr_t)&Bs[p][sr][sc];
    const unsigned long long ga = (unsigned long long)(uintptr_t)(aptr + k0);
    const unsigned long long gb = (unsigned long long)(uintptr_t)(wptr + k0);
    asm volatile("global_load_async_to_lds_b128 %0, %1, off"
                 :: "v"(la), "v"(ga) : "memory");
    asm volatile("global_load_async_to_lds_b128 %0, %1, off offset:16"
                 :: "v"(la), "v"(ga) : "memory");
    asm volatile("global_load_async_to_lds_b128 %0, %1, off"
                 :: "v"(lb), "v"(gb) : "memory");
    asm volatile("global_load_async_to_lds_b128 %0, %1, off offset:16"
                 :: "v"(lb), "v"(gb) : "memory");
  };

  stage_async(0, 0);
  asm volatile("s_wait_asynccnt 0x0" ::: "memory");
  __syncthreads();

  int p = 0;
  for (int k0 = 0; k0 < K; k0 += BK, p ^= 1) {
    const bool more = (k0 + BK) < K;
    if (more) stage_async(k0 + BK, p ^ 1);   // DMA overlaps the WMMAs below

    v16bf afr[2], bfr[4];
    #pragma unroll
    for (int mi = 0; mi < 2; ++mi)    // A lane: K = 8*lhalf+i, 8*lhalf+16+i
      afr[mi] = ld_frag(&As[p][wm * 32 + mi * 16 + l16][0], lhalf, lhalf + 2);
    #pragma unroll
    for (int ni = 0; ni < 4; ++ni)    // B lane: K = 8*lhalf+i, 8*lhalf+16+i
      bfr[ni] = ld_frag(&Bs[p][wn * 64 + ni * 16 + l16][0], lhalf, lhalf + 2);

    #pragma unroll
    for (int mi = 0; mi < 2; ++mi)
      #pragma unroll
      for (int ni = 0; ni < 4; ++ni)
        acc[mi][ni] = __builtin_amdgcn_wmma_f32_16x16x32_bf16(
            false, afr[mi], false, bfr[ni], (short)0, acc[mi][ni], false, false);

    if (more) {
      asm volatile("s_wait_asynccnt 0x0" ::: "memory");
      __syncthreads();                // single barrier per K-step
    }
  }

  // epilogue: C/D layout row = v + 8*lhalf, col = l16
  #pragma unroll
  for (int mi = 0; mi < 2; ++mi) {
    #pragma unroll
    for (int ni = 0; ni < 4; ++ni) {
      const int gcol = n0 + wn * 64 + ni * 16 + l16;
      if (NGUARD && gcol >= N) continue;
      #pragma unroll
      for (int v = 0; v < 8; ++v) {
        const int grow = m0 + wm * 32 + mi * 16 + lhalf * 8 + v;
        float val = acc[mi][ni][v];
        if (EPI == 1)      val += bias[gcol];
        else if (EPI == 2) val = softplusf(val + bias[gcol]);
        else if (EPI == 3) val += resid[(size_t)grow * ldc + gcol];
        else if (EPI == 4) val += bias[gcol] + resid[(size_t)grow * ldc + gcol];
        C[(size_t)grow * ldc + gcol] = val;
      }
    }
  }
}

// ---------------------------------------------------------------------------
// Weight transpose + f32->bf16: W[K,N](ldin) -> Wt[NP,K], rows >= N zero-fill.
// LDS-tiled so both global sides stay coalesced.  Block 256 = 32x8.
// ---------------------------------------------------------------------------
__global__ __launch_bounds__(256)
void transpose_cvt_kernel(const float* __restrict__ in, __bf16* __restrict__ out,
                          int K, int N, int ldin) {
  __shared__ float tile[32][33];
  const int kb = blockIdx.x * 32;
  const int nb = blockIdx.y * 32;
  const int tx = threadIdx.x & 31;
  const int ty = threadIdx.x >> 5;    // 0..7
  #pragma unroll
  for (int j = 0; j < 4; ++j) {
    const int n = nb + tx;
    tile[ty + j * 8][tx] =
        (n < N) ? in[(size_t)(kb + ty + j * 8) * ldin + n] : 0.f;
  }
  __syncthreads();
  #pragma unroll
  for (int j = 0; j < 4; ++j)
    out[(size_t)(nb + ty + j * 8) * K + (kb + tx)] = (__bf16)tile[tx][ty + j * 8];
}

// ---------------------------------------------------------------------------
// f32 -> bf16 (packed out), strided input rows (for dt_lo slice).
// ---------------------------------------------------------------------------
__global__ __launch_bounds__(256)
void cvt2d_bf16_kernel(const float* __restrict__ in, __bf16* __restrict__ out,
                       long rows, long cols, long ldin) {
  const long i4 = (long)blockIdx.x * 256 + threadIdx.x;
  if (i4 >= rows * cols / 4) return;
  const long e = i4 * 4;
  const long r = e / cols, c = e - r * cols;
  const float4 v = *(const float4*)&in[r * ldin + c];
  union { __bf16 h[4]; uint2 u; } p;
  p.h[0] = (__bf16)v.x; p.h[1] = (__bf16)v.y;
  p.h[2] = (__bf16)v.z; p.h[3] = (__bf16)v.w;
  *(uint2*)&out[e] = p.u;
}

// ---------------------------------------------------------------------------
// LayerNorm over DM=1024, one block per token row, bf16 output (GEMM feed).
// ---------------------------------------------------------------------------
__global__ __launch_bounds__(256)
void layernorm_kernel(const float* __restrict__ x, const float* __restrict__ g,
                      const float* __restrict__ b, __bf16* __restrict__ out) {
  const int row = blockIdx.x;
  const float* xr = x + (size_t)row * DM;
  float s1 = 0.f, s2 = 0.f;
  for (int i = threadIdx.x; i < DM; i += 256) { float v = xr[i]; s1 += v; s2 += v * v; }
  __shared__ float r1[256], r2[256];
  r1[threadIdx.x] = s1; r2[threadIdx.x] = s2;
  __syncthreads();
  for (int off = 128; off > 0; off >>= 1) {
    if (threadIdx.x < off) {
      r1[threadIdx.x] += r1[threadIdx.x + off];
      r2[threadIdx.x] += r2[threadIdx.x + off];
    }
    __syncthreads();
  }
  const float mu   = r1[0] * (1.f / DM);
  const float var  = r2[0] * (1.f / DM) - mu * mu;
  const float rstd = rsqrtf(var + 1e-5f);
  for (int i = threadIdx.x; i < DM; i += 256)
    out[(size_t)row * DM + i] = (__bf16)((xr[i] - mu) * rstd * g[i] + b[i]);
}

// ---------------------------------------------------------------------------
// Depthwise causal conv1d (w=4) + bias + SiLU; writes f32 (scan) + bf16 (GEMM).
// ---------------------------------------------------------------------------
__global__ __launch_bounds__(256)
void conv_silu_kernel(const float* __restrict__ xz, const float* __restrict__ cw,
                      const float* __restrict__ cb, float* __restrict__ out,
                      __bf16* __restrict__ out_h) {
  const size_t idx = (size_t)blockIdx.x * blockDim.x + threadIdx.x;
  if (idx >= (size_t)M_ROWS * DI) return;
  const int d = (int)(idx % DI);
  const int r = (int)(idx / DI);
  const int l = r % SEQ;
  float acc = cb[d];
  #pragma unroll
  for (int j = 0; j < DCONV; ++j) {
    const int ls = l - (DCONV - 1) + j;
    if (ls >= 0)
      acc += cw[d * DCONV + j] * xz[(size_t)(r - (DCONV - 1) + j) * (2 * DI) + d];
  }
  const float v = siluf(acc);
  out[idx]   = v;
  out_h[idx] = (__bf16)v;
}

// ---------------------------------------------------------------------------
// Selective scan: ONE LANE PER (b, d, state n).  65536 threads = 2048 waves.
// Per step: 1 exp per lane; 16-state dot with C via 4x shfl_xor (wave32).
// ---------------------------------------------------------------------------
__global__ __launch_bounds__(256)
void scan_kernel(const float* __restrict__ dt, const float* __restrict__ dbc,
                 const float* __restrict__ xc, const float* __restrict__ xz,
                 const float* __restrict__ a_log, const float* __restrict__ d_skip,
                 __bf16* __restrict__ y) {
  const int t  = blockIdx.x * blockDim.x + threadIdx.x;   // 0 .. B*DI*DS-1
  if (t >= BATCH * DI * DS) return;
  const int n  = t & (DS - 1);        // state index = lane group position
  const int ch = t >> 4;              // channel = b*DI + d
  const int b  = ch / DI, d = ch % DI;

  const float a   = -__expf(a_log[d * DS + n]);
  const float dsk = d_skip[d];
  float s = 0.f;

  for (int l = 0; l < SEQ; ++l) {
    const size_t r  = (size_t)b * SEQ + l;
    const float dtv = dt[r * DI + d];                       // broadcast in group
    const float xv  = xc[r * DI + d];                       // broadcast in group
    const float Bn  = dbc[r * (DTR + 2 * DS) + DTR + n];
    const float Cn  = dbc[r * (DTR + 2 * DS) + DTR + DS + n];
    s = s * __expf(dtv * a) + dtv * xv * Bn;
    float part = s * Cn;
    part += __shfl_xor(part, 1);
    part += __shfl_xor(part, 2);
    part += __shfl_xor(part, 4);
    part += __shfl_xor(part, 8);                            // sum over 16 states
    if (n == 0) {
      const float zv = xz[r * (2 * DI) + DI + d];
      y[r * DI + d] = (__bf16)((part + dsk * xv) * siluf(zv));
    }
  }
}

// ---------------------------------------------------------------------------
// GeGLU gate -> bf16 (GEMM feed).
// ---------------------------------------------------------------------------
__global__ __launch_bounds__(256)
void geglu_kernel(const float* __restrict__ p, __bf16* __restrict__ out) {
  const size_t idx = (size_t)blockIdx.x * blockDim.x + threadIdx.x;
  if (idx >= (size_t)M_ROWS * DFF) return;
  const size_t row = idx / DFF;
  const int    col = (int)(idx % DFF);
  const float av = p[row * (2 * DFF) + col];
  const float gv = p[row * (2 * DFF) + DFF + col];
  out[idx] = (__bf16)(av * geluf(gv));
}

// ---------------------------------------------------------------------------
// Workspace arena.  f32 region, then bf16 region; FF phase overlays dead bufs.
// ---------------------------------------------------------------------------
// f32 (element offsets)
constexpr size_t F_X2  = 0;
constexpr size_t F_XZ  = F_X2  + (size_t)M_ROWS * DM;
constexpr size_t F_XC  = F_XZ  + (size_t)M_ROWS * 2 * DI;
constexpr size_t F_DBC = F_XC  + (size_t)M_ROWS * DI;
constexpr size_t F_DTB = F_DBC + (size_t)M_ROWS * (DTR + 2 * DS);
constexpr size_t F_END = F_DTB + (size_t)M_ROWS * DI;
constexpr size_t F_PB  = F_XZ;                                     // ff1 out (reuse)
// bf16 (element offsets; transposed weights Wt[NP][K], NP padded to 128)
constexpr size_t H_WIN  = 0;                                       // 4096 x 1024
constexpr size_t H_WX   = H_WIN  + (size_t)(2 * DI) * DM;          // 128 x 2048
constexpr size_t H_WDT  = H_WX   + (size_t)128 * DI;               // 2048 x 64
constexpr size_t H_WOUT = H_WDT  + (size_t)DI * DTR;               // 1024 x 2048
constexpr size_t H_WFF1 = H_WOUT + (size_t)DM * DI;                // 8192 x 1024
constexpr size_t H_WFF2 = H_WFF1 + (size_t)(2 * DFF) * DM;         // 1024 x 4096
constexpr size_t H_H    = H_WFF2 + (size_t)DM * DFF;
constexpr size_t H_XC   = H_H    + (size_t)M_ROWS * DM;
constexpr size_t H_DTLO = H_XC   + (size_t)M_ROWS * DI;
constexpr size_t H_Y    = H_DTLO + (size_t)M_ROWS * DTR;
constexpr size_t H_FA   = H_XC;                                    // geglu out (reuse)

extern "C" void kernel_launch(void* const* d_in, const int* in_sizes, int n_in,
                              void* d_out, int out_size, void* d_ws, size_t ws_size,
                              hipStream_t stream) {
  (void)in_sizes; (void)n_in; (void)out_size; (void)ws_size;

  const float* x      = (const float*)d_in[0];
  const float* ln1_g  = (const float*)d_in[1];
  const float* ln1_b  = (const float*)d_in[2];
  const float* w_in   = (const float*)d_in[3];
  const float* conv_w = (const float*)d_in[4];
  const float* conv_b = (const float*)d_in[5];
  const float* w_x    = (const float*)d_in[6];
  const float* w_dt   = (const float*)d_in[7];
  const float* b_dt   = (const float*)d_in[8];
  const float* a_log  = (const float*)d_in[9];
  const float* d_skip = (const float*)d_in[10];
  const float* w_out  = (const float*)d_in[11];
  const float* ln2_g  = (const float*)d_in[12];
  const float* ln2_b  = (const float*)d_in[13];
  const float* w_ff1  = (const float*)d_in[14];
  const float* b_ff1  = (const float*)d_in[15];
  const float* w_ff2  = (const float*)d_in[16];
  const float* b_ff2  = (const float*)d_in[17];

  float*  wf = (float*)d_ws;
  __bf16* wh = (__bf16*)((char*)d_ws + F_END * sizeof(float));

  float* x2  = wf + F_X2;
  float* xz  = wf + F_XZ;
  float* xc  = wf + F_XC;
  float* dbc = wf + F_DBC;
  float* dtb = wf + F_DTB;
  float* pb  = wf + F_PB;

  __bf16* w_in_t  = wh + H_WIN;
  __bf16* w_x_t   = wh + H_WX;
  __bf16* w_dt_t  = wh + H_WDT;
  __bf16* w_out_t = wh + H_WOUT;
  __bf16* w_ff1_t = wh + H_WFF1;
  __bf16* w_ff2_t = wh + H_WFF2;
  __bf16* h_h     = wh + H_H;
  __bf16* xc_h    = wh + H_XC;
  __bf16* dtlo_h  = wh + H_DTLO;
  __bf16* y_h     = wh + H_Y;
  __bf16* fa_h    = wh + H_FA;

  float* out = (float*)d_out;

  const dim3 blk(256);
  auto gg = [](int Mr, int Nc) { return dim3((Nc + 127) / 128, (Mr + 127) / 128); };
  auto tp = [&](const float* w, __bf16* o, int K, int N, int NP) {
    transpose_cvt_kernel<<<dim3(K / 32, NP / 32), blk, 0, stream>>>(w, o, K, N, N);
  };

  // ---- weight transpose + bf16 convert (once per call) ----
  tp(w_in,  w_in_t,  DM,  2 * DI,       2 * DI);
  tp(w_x,   w_x_t,   DI,  DTR + 2 * DS, 128);
  tp(w_dt,  w_dt_t,  DTR, DI,           DI);
  tp(w_out, w_out_t, DI,  DM,           DM);
  tp(w_ff1, w_ff1_t, DM,  2 * DFF,      2 * DFF);
  tp(w_ff2, w_ff2_t, DFF, DM,           DM);

  // ---- mamba branch ----
  layernorm_kernel<<<M_ROWS, blk, 0, stream>>>(x, ln1_g, ln1_b, h_h);
  gemm_bf16_wmma<0, false><<<gg(M_ROWS, 2 * DI), blk, 0, stream>>>(
      h_h, w_in_t, nullptr, nullptr, xz, M_ROWS, 2 * DI, DM, DM, 2 * DI);
  conv_silu_kernel<<<(M_ROWS * DI) / 256, blk, 0, stream>>>(xz, conv_w, conv_b, xc, xc_h);
  gemm_bf16_wmma<0, true><<<gg(M_ROWS, DTR + 2 * DS), blk, 0, stream>>>(
      xc_h, w_x_t, nullptr, nullptr, dbc, M_ROWS, DTR + 2 * DS, DI, DI, DTR + 2 * DS);
  cvt2d_bf16_kernel<<<(M_ROWS * DTR / 4 + 255) / 256, blk, 0, stream>>>(
      dbc, dtlo_h, M_ROWS, DTR, DTR + 2 * DS);       // dt_lo slice -> packed bf16
  gemm_bf16_wmma<2, false><<<gg(M_ROWS, DI), blk, 0, stream>>>(
      dtlo_h, w_dt_t, b_dt, nullptr, dtb, M_ROWS, DI, DTR, DTR, DI);
  scan_kernel<<<(BATCH * DI * DS) / 256, blk, 0, stream>>>(
      dtb, dbc, xc, xz, a_log, d_skip, y_h);
  gemm_bf16_wmma<3, false><<<gg(M_ROWS, DM), blk, 0, stream>>>(
      y_h, w_out_t, nullptr, x, x2, M_ROWS, DM, DI, DI, DM);

  // ---- GeGLU FF branch ----
  layernorm_kernel<<<M_ROWS, blk, 0, stream>>>(x2, ln2_g, ln2_b, h_h);
  gemm_bf16_wmma<1, false><<<gg(M_ROWS, 2 * DFF), blk, 0, stream>>>(
      h_h, w_ff1_t, b_ff1, nullptr, pb, M_ROWS, 2 * DFF, DM, DM, 2 * DFF);
  geglu_kernel<<<(M_ROWS * DFF) / 256, blk, 0, stream>>>(pb, fa_h);
  gemm_bf16_wmma<4, false><<<gg(M_ROWS, DM), blk, 0, stream>>>(
      fa_h, w_ff2_t, b_ff2, x2, out, M_ROWS, DM, DFF, DFF, DM);
}
